// PPGDPOPolicy_40862318854411
// MI455X (gfx1250) — matrix operand order repaired and checked by reading, not compiled, not run on hardware
//
#include <hip/hip_runtime.h>
#include <math.h>

typedef __attribute__((ext_vector_type(16))) _Float16 v16h;
typedef __attribute__((ext_vector_type(8)))  _Float16 v8h;
typedef __attribute__((ext_vector_type(8)))  float    v8f;

// ---- constants from the reference ----
#define MSTEPS   60
#define NPTS     32
#define BPATH    4096           // RPTS * NPTS
#define DT_C     0.025f         // 1.5 / 60
#define SQDT_C   0.15811388300841897f
#define SIGMA_C  0.2f
#define SGA_C    0.16f          // SIGMA * ALPHA (0.2*0.8)
#define RHO_C    0.3f
#define SQ1MR2_C 0.9539392014169456f
#define KAPPA_C  1.2f
#define SIGY_C   0.3f
#define RRATE_C  0.02f
#define LOGLB_C  -6.907755278982137f   // log(1e-3)
#define QEXP_C   -4.0f          // 1 - GAMMA

__global__ void __launch_bounds__(128) ppgdpo_zero_acc(float* acc) {
    int i = threadIdx.x;
    if (i < 96) acc[i] = 0.0f;
}

// One wave (32 threads) per block; each wave owns 16 paths (a 16-row WMMA tile).
// blockIdx.x in [0,256): path tile.  blockIdx.y in {0,1}: antithetic branch.
__global__ void __launch_bounds__(32)
ppgdpo_sim_kernel(const float* __restrict__ W0g, const float* __restrict__ Y0g,
                  const float* __restrict__ noise,
                  const float* __restrict__ w1, const float* __restrict__ b1,
                  const float* __restrict__ w2, const float* __restrict__ b2,
                  const float* __restrict__ w3, const float* __restrict__ b3,
                  float* __restrict__ acc)
{
    __shared__ __attribute__((aligned(16))) float    U2[5][16][64];   // pre-acts / post-acts layer2
    __shared__ __attribute__((aligned(16))) _Float16 hA[5][16][64];   // f16 A-operand rows for layer2
    __shared__ __attribute__((aligned(16))) _Float16 w2h[64 * 64];    // f16 copy of w2
    __shared__ float xb[16][8];                                       // per-path policy inputs + tangents
    __shared__ float w1s[192], b1s[64], b2s[64], w3s[64];
    __shared__ float b3s;

    const int lane  = threadIdx.x;
    const int pbase = blockIdx.x * 16;
    const int sim   = blockIdx.y;          // 0: anti=+1, 1: anti=-1
    const float anti = (sim == 0) ? 1.0f : -1.0f;

    // Stage weights into LDS (f16 for the WMMA layer).
    for (int i = lane; i < 64 * 64; i += 32) w2h[i] = (_Float16)w2[i];
    for (int i = lane; i < 192;     i += 32) w1s[i] = w1[i];
    for (int i = lane; i < 64;      i += 32) { b1s[i] = b1[i]; b2s[i] = b2[i]; w3s[i] = w3[i]; }
    if (lane == 0) b3s = b3[0];
    __syncthreads();

    // Preload the 8 B-operand fragments of w2 (held in VGPRs for all 60 steps).
    // 16-bit B 32x16, this code's contiguous k order (same as A): lanes 0-15 hold K=0..15, lanes 16-31 hold K=16..31, N = lane%16.
    const int col  = lane & 15;
    const int koff = (lane < 16) ? 0 : 16;
    v16h bf[4][2];
#pragma unroll
    for (int nt = 0; nt < 4; ++nt)
#pragma unroll
        for (int kh = 0; kh < 2; ++kh)
#pragma unroll
            for (int e = 0; e < 16; ++e)
                bf[nt][kh][e] = w2h[(kh * 32 + koff + e) * 64 + nt * 16 + col];

    // ---- per-path state (lanes 0..15), forward-over-forward tangents ----
    float logW = 0.f, tW = 0.f, tY = 0.f, sWW = 0.f, sWY = 0.f, Yv = 0.f;
    if (lane < 16) {
        const int p  = pbase + lane;
        const float W0  = W0g[p & 31];
        const float Y0v = Y0g[p & 31];
        const float Wc  = fmaxf(W0, 1e-3f);
        logW = logf(Wc);
        tW   = (W0 > 1e-3f) ? 1.0f / W0 : 0.0f;            // d logW / dW0
        sWW  = (W0 > 1e-3f) ? -1.0f / (W0 * W0) : 0.0f;    // d2 logW / dW0^2
        Yv   = Y0v;
    }
    float eY  = 1.0f;     // dY_t/dY0 = (1-kappa*dt)^t, uniform across lanes
    float TmT = 1.5f;

    const float* z1b = noise + (size_t)(sim * 2 + 0) * MSTEPS * BPATH;
    const float* z2b = noise + (size_t)(sim * 2 + 1) * MSTEPS * BPATH;
    const int m16 = lane & 15;

    for (int step = 0; step < MSTEPS; ++step) {
        // Publish policy inputs + tangents (W = exp(logW); chain rule through exp).
        if (lane < 16) {
            const float Wv = expf(logW);
            xb[lane][0] = Wv;                      // W
            xb[lane][1] = TmT;                     // T - t
            xb[lane][2] = Yv;                      // Y
            xb[lane][3] = Wv * tW;                 // dW/dW0
            xb[lane][4] = Wv * tY;                 // dW/dY0
            xb[lane][5] = Wv * (tW * tW + sWW);    // d2W/dW0^2
            xb[lane][6] = Wv * (tW * tY + sWY);    // d2W/dW0dY0
        }
        __syncthreads();

        // Layer 1 (K=3, VALU) + tanh forward-over-forward combine -> f16 A rows.
        for (int idx = lane; idx < 16 * 64; idx += 32) {
            const int m = idx >> 6, j = idx & 63;
            const float wj0 = w1s[j], wj1 = w1s[64 + j], wj2 = w1s[128 + j];
            const float u0  = xb[m][0] * wj0 + xb[m][1] * wj1 + xb[m][2] * wj2 + b1s[j];
            const float ua  = xb[m][3] * wj0;
            const float ub  = xb[m][4] * wj0 + eY * wj2;
            const float uaa = xb[m][5] * wj0;
            const float uab = xb[m][6] * wj0;
            const float y  = tanhf(u0);
            const float s  = 1.0f - y * y;
            const float ya = s * ua, yb = s * ub;
            const float yaa = s * uaa - 2.0f * y * ya * ua;
            const float yab = s * uab - 2.0f * y * yb * ua;
            hA[0][m][j] = (_Float16)y;
            hA[1][m][j] = (_Float16)ya;
            hA[2][m][j] = (_Float16)yb;
            hA[3][m][j] = (_Float16)yaa;
            hA[4][m][j] = (_Float16)yab;
        }
        __syncthreads();

        // Layer 2: five 16x64 @ 64x64 GEMMs on the matrix pipe.
        // 16-bit A 16x32, this code's contiguous k order (same as B): lanes 0-15 hold K=0..15, lanes 16-31 hold K=16..31.
        {
            const int aoff = (lane < 16) ? 0 : 16;
            const int mrow = (lane < 16) ? 0 : 8;
#pragma unroll
            for (int q = 0; q < 5; ++q) {
                const _Float16* row = &hA[q][m16][0];
                const v8h c00 = *(const v8h*)(row + aoff);
                const v8h c01 = *(const v8h*)(row + aoff + 8);
                const v8h c10 = *(const v8h*)(row + 32 + aoff);
                const v8h c11 = *(const v8h*)(row + 32 + aoff + 8);
                const v16h a0 = __builtin_shufflevector(c00, c01, 0,1,2,3,4,5,6,7,8,9,10,11,12,13,14,15);
                const v16h a1 = __builtin_shufflevector(c10, c11, 0,1,2,3,4,5,6,7,8,9,10,11,12,13,14,15);
#pragma unroll
                for (int nt = 0; nt < 4; ++nt) {
                    v8f c = {};
                    c = __builtin_amdgcn_wmma_f32_16x16x32_f16(false, a0, false, bf[nt][0],
                                                               (short)0, c, false, false);
                    c = __builtin_amdgcn_wmma_f32_16x16x32_f16(false, a1, false, bf[nt][1],
                                                               (short)0, c, false, false);
#pragma unroll
                    for (int g = 0; g < 8; ++g)
                        U2[q][mrow + g][nt * 16 + col] = c[g];  // C layout -> (m,j)
                }
            }
        }
        __syncthreads();

        // Layer 2 tanh combine (elementwise, in place; +b2 on the value).
        for (int idx = lane; idx < 16 * 64; idx += 32) {
            const int m = idx >> 6, j = idx & 63;
            const float u0  = U2[0][m][j] + b2s[j];
            const float ua  = U2[1][m][j];
            const float ub  = U2[2][m][j];
            const float uaa = U2[3][m][j];
            const float uab = U2[4][m][j];
            const float y  = tanhf(u0);
            const float s  = 1.0f - y * y;
            const float ya = s * ua, yb = s * ub;
            const float yaa = s * uaa - 2.0f * y * ya * ua;
            const float yab = s * uab - 2.0f * y * yb * ua;
            U2[0][m][j] = y;  U2[1][m][j] = ya;  U2[2][m][j] = yb;
            U2[3][m][j] = yaa; U2[4][m][j] = yab;
        }
        __syncthreads();

        // Layer 3 (N=1 dot) + SDE update with tangent propagation (lane = path).
        if (lane < 16) {
            float pi = b3s, pa = 0.f, pb = 0.f, paa = 0.f, pab = 0.f;
            for (int j = 0; j < 64; ++j) {
                const float w = w3s[j];
                pi  += U2[0][lane][j] * w;
                pa  += U2[1][lane][j] * w;
                pb  += U2[2][lane][j] * w;
                paa += U2[3][lane][j] * w;
                pab += U2[4][lane][j] * w;
            }
            const int p  = pbase + lane;
            const float z1 = z1b[step * BPATH + p];
            const float z2 = z2b[step * BPATH + p];
            const float zW = anti * z1;
            const float zY = anti * (RHO_C * z1 + SQ1MR2_C * z2);

            // logW' = logW + (r + pi*sga*Y - 0.5 pi^2 s^2) dt + pi*s*sqdt*zW
            const float Fp  = SGA_C * Yv * DT_C - pi * SIGMA_C * SIGMA_C * DT_C + SIGMA_C * SQDT_C * zW;
            const float Fy  = pi * SGA_C * DT_C;
            const float Fpp = -SIGMA_C * SIGMA_C * DT_C;
            const float Fpy = SGA_C * DT_C;

            logW += (RRATE_C + pi * SGA_C * Yv - 0.5f * pi * pi * SIGMA_C * SIGMA_C) * DT_C
                    + pi * SIGMA_C * SQDT_C * zW;
            const float ntW = tW  + Fp * pa;
            const float ntY = tY  + Fp * pb + Fy * eY;
            const float nWW = sWW + Fpp * pa * pa + Fp * paa;
            const float nWY = sWY + Fpp * pa * pb + Fpy * pa * eY + Fp * pab;
            tW = ntW; tY = ntY; sWW = nWW; sWY = nWY;

            Yv = Yv + KAPPA_C * (0.0f - Yv) * DT_C + SIGY_C * SQDT_C * zY;

            // clip(W, 1e-3): clamp logW and cut tangents when active.
            if (logW < LOGLB_C) { logW = LOGLB_C; tW = 0.f; tY = 0.f; sWW = 0.f; sWY = 0.f; }
        }
        eY  *= (1.0f - KAPPA_C * DT_C);
        TmT -= DT_C;
        __syncthreads();
    }

    // U = W^(1-g)/(1-g);  E = exp(q*logW), q = -4.
    if (lane < 16) {
        const float E   = expf(QEXP_C * logW);
        const float Ua  = E * tW;                                  // dU/dW0
        const float Uaa = E * (QEXP_C * tW * tW + sWW);            // d2U/dW0^2
        const float Uab = E * (QEXP_C * tW * tY + sWY);            // d2U/dW0dY0
        const int pt = (pbase + lane) & 31;
        atomicAdd(&acc[pt],       0.5f * Ua);    // antithetic 0.5 factor per branch
        atomicAdd(&acc[32 + pt],  0.5f * Uaa);
        atomicAdd(&acc[64 + pt],  0.5f * Uab);
    }
}

__global__ void __launch_bounds__(32)
ppgdpo_project_kernel(const float* __restrict__ W, const float* __restrict__ Y,
                      const float* __restrict__ acc, float* __restrict__ out)
{
    const int n = threadIdx.x;
    if (n >= NPTS) return;
    // Exact reference scaling: lam = sum(Ua)/ (RPTS^2 * REPEATS); d* = sum / (RPTS^3 * REPEATS)
    const float SL = 1.0f / (128.0f * 128.0f * 128.0f);
    const float SD = SL / 128.0f;
    const float lam = acc[n]      * SL;
    const float dW  = acc[32 + n] * SD;
    const float dY  = acc[64 + n] * SD;
    const float Wv = W[n], Yv = Y[n];
    const float mu_minus_r = SIGMA_C * (0.8f * Yv);
    const float coeff  = -1.0f / (Wv * dW + 1e-8f);
    const float inv_s2 = 1.0f / (SIGMA_C * SIGMA_C);
    const float myo    = coeff * (lam * mu_minus_r * inv_s2);
    const float hedge  = coeff * (SIGMA_C * RHO_C * SIGY_C * dY * inv_s2);
    const float v = myo + hedge;
    out[n] = fminf(fmaxf(v, -2.0f), 2.0f);
}

extern "C" void kernel_launch(void* const* d_in, const int* in_sizes, int n_in,
                              void* d_out, int out_size, void* d_ws, size_t ws_size,
                              hipStream_t stream) {
    const float* W     = (const float*)d_in[0];
    const float* TmT   = (const float*)d_in[1];  (void)TmT;  // constant T_H, baked in
    const float* Y     = (const float*)d_in[2];
    const float* noise = (const float*)d_in[3];
    const float* w1    = (const float*)d_in[4];
    const float* b1    = (const float*)d_in[5];
    const float* w2    = (const float*)d_in[6];
    const float* b2    = (const float*)d_in[7];
    const float* w3    = (const float*)d_in[8];
    const float* b3    = (const float*)d_in[9];
    (void)in_sizes; (void)n_in; (void)out_size; (void)ws_size;

    float* acc = (float*)d_ws;   // 96 floats: [lam | d2W | d2Y] x 32 points

    ppgdpo_zero_acc<<<1, 128, 0, stream>>>(acc);
    dim3 grid(BPATH / 16, 2);    // 256 path tiles x 2 antithetic branches
    ppgdpo_sim_kernel<<<grid, 32, 0, stream>>>(W, Y, noise, w1, b1, w2, b2, w3, b3, acc);
    ppgdpo_project_kernel<<<1, 32, 0, stream>>>(W, Y, acc, (float*)d_out);
}
